// G2ANet_Actor_83167746719906
// MI455X (gfx1250) — matrix operand order repaired and checked by reading, not compiled, not run on hardware
//
#include <hip/hip_runtime.h>
#include <hip/hip_bf16.h>
#include <math.h>

typedef __attribute__((ext_vector_type(16))) _Float16 v16h;
typedef __attribute__((ext_vector_type(8)))  _Float16 v8h;
typedef __attribute__((ext_vector_type(8)))  float    v8f;

#define B_   64
#define N_   64
#define H_   128
#define A_   16
#define BN_  4096
#define G3_  384
#define PLD  1792
// P (fused projection) column offsets
#define OFF_Q  0
#define OFF_K  128
#define OFF_FS 256
#define OFF_FO 640
#define OFF_BS 1024
#define OFF_BO 1408

__device__ __forceinline__ float sigmoidf_(float x) { return 1.f / (1.f + expf(-x)); }

__device__ __forceinline__ v8f wmma16x16x32(v16h a, v16h b, v8f c) {
  // D = A(16x32 f16) * B(32x16 f16) + C(16x16 f32)
  return __builtin_amdgcn_wmma_f32_16x16x32_f16(false, a, false, b, (short)0, c, false, false);
}

// ---------------- f32 -> f16 submatrix convert ----------------
__global__ void cvt_f32_f16(const float* __restrict__ src, int sld,
                            _Float16* __restrict__ dst, int dld, int rows, int cols) {
  int i = blockIdx.x * blockDim.x + threadIdx.x;
  if (i >= rows * cols) return;
  int r = i / cols, c = i - r * cols;
  dst[(size_t)r * dld + c] = (_Float16)src[(size_t)r * sld + c];
}

// hprev = lstm_h * masks, converted into A2[:,128:256]
__global__ void hprev_cvt(const float* __restrict__ lstm_h, const float* __restrict__ masks,
                          _Float16* __restrict__ dst, int dld) {
  int i = blockIdx.x * blockDim.x + threadIdx.x;
  if (i >= BN_ * H_) return;
  int r = i >> 7, c = i & 127;
  dst[(size_t)r * dld + 128 + c] = (_Float16)(lstm_h[i] * masks[r]);
}

// ---------------- generic WMMA GEMM: C = act(A @ W^T + b1 + b2) ----------------
// A: MxK f16 row-major (lda), W: NxK f16 row-major (ldw). One wave per 16x16 tile.
__global__ void __launch_bounds__(256) gemm_wmma(
    const _Float16* __restrict__ A, int lda,
    const _Float16* __restrict__ W, int ldw,
    const float* __restrict__ bias1, const float* __restrict__ bias2,
    float* __restrict__ C32, int ldc32,
    _Float16* __restrict__ C16, int ldc16,
    int M, int N, int K, int relu) {
  int wid  = (blockIdx.x * blockDim.x + threadIdx.x) >> 5;
  int lane = threadIdx.x & 31;
  int ntile = N >> 4;
  int total = (M >> 4) * ntile;
  if (wid >= total) return;               // whole-wave uniform exit
  int tm = wid / ntile, tn = wid - tm * ntile;
  int rb = tm << 4, cb = tn << 4;
  int g = lane >> 4, ln = lane & 15;
  v8f acc;
#pragma unroll
  for (int e = 0; e < 8; ++e) acc[e] = 0.f;
  for (int kb = 0; kb < K; kb += 32) {
    // A layout: lane ln holds row rb+ln; halves e -> K=kb+16g+e (contiguous)
    v8h a0 = *(const v8h*)(A + (size_t)(rb + ln) * lda + kb + 16 * g);
    v8h a1 = *(const v8h*)(A + (size_t)(rb + ln) * lda + kb + 16 * g + 8);
    v16h a;
#pragma unroll
    for (int e = 0; e < 8; ++e) { a[e] = a0[e]; a[e + 8] = a1[e]; }
    // B layout: lane ln -> col cb+ln, halves e -> K=kb+16g+e (contiguous in W row)
    v16h bm = *(const v16h*)(W + (size_t)(cb + ln) * ldw + kb + 16 * g);
    acc = wmma16x16x32(a, bm, acc);
  }
#pragma unroll
  for (int v = 0; v < 8; ++v) {
    int r = rb + v + 8 * g, c = cb + ln;
    float val = acc[v];
    if (bias1) val += bias1[c];
    if (bias2) val += bias2[c];
    if (relu) val = fmaxf(val, 0.f);
    if (C32) C32[(size_t)r * ldc32 + c] = val;
    if (C16) C16[(size_t)r * ldc16 + c] = (_Float16)val;
  }
}

// ---------------- LSTM elementwise ----------------
__global__ void lstm_ew(const float* __restrict__ gates, const float* __restrict__ lstm_c,
                        const float* __restrict__ masks, float* __restrict__ hout,
                        float* __restrict__ cout, _Float16* __restrict__ h16) {
  int i = blockIdx.x * blockDim.x + threadIdx.x;
  if (i >= BN_ * H_) return;
  int r = i >> 7, j = i & 127;
  const float* gr = gates + (size_t)r * 512;
  float ig = gr[j], fg = gr[128 + j], gg = gr[256 + j], og = gr[384 + j];
  float cp = lstm_c[i] * masks[r];
  float cn = sigmoidf_(fg) * cp + sigmoidf_(ig) * tanhf(gg);
  float hv = sigmoidf_(og) * tanhf(cn);
  hout[i] = hv; cout[i] = cn; h16[i] = (_Float16)hv;
}

// ---------------- bidirectional GRU scan (WMMA recurrence in LDS) ----------------
// blocks 0..255: forward over rows [b*16, b*16+16); blocks 256..511: backward.
__global__ void __launch_bounds__(256) gru_kernel(
    const float* __restrict__ P,
    const _Float16* __restrict__ whhF, const _Float16* __restrict__ whhB,
    const float* __restrict__ bihF, const float* __restrict__ bhhF,
    const float* __restrict__ bihB, const float* __restrict__ bhhB,
    const float* __restrict__ W_hard,
    float* __restrict__ logF, float* __restrict__ logB) {
  __shared__ __align__(32) _Float16 sWhh[G3_ * H_];   // 96 KB  recurrent weights
  __shared__ __align__(32) _Float16 sH16[16 * H_];    // 4 KB   state (f16 for WMMA A)
  __shared__ float sH32[16 * H_];                     // 8 KB   state (f32)
  __shared__ float sG[16 * G3_];                      // 24 KB  pre-activations (gh for n-gate)
  __shared__ float sGin[16 * H_];                     // 8 KB   gi for n-gate
  __shared__ float sL[32];                            // per-row W_hard logit accum

  int tid = threadIdx.x;
  int dir = (blockIdx.x >= 256) ? 1 : 0;
  int rb  = (blockIdx.x & 255) << 4;
  const _Float16* whh = dir ? whhB : whhF;
  const float* bih = dir ? bihB : bihF;
  const float* bhh = dir ? bhhB : bhhF;
  float* lout = dir ? logB : logF;
  int selfOff  = dir ? OFF_BS : OFF_FS;
  int otherOff = dir ? OFF_BO : OFF_FO;
  int hOff = dir ? 128 : 0;

  { // stage Whh into LDS (coalesced dwords)
    const unsigned int* wsrc = (const unsigned int*)whh;
    unsigned int* wdst = (unsigned int*)sWhh;
    for (int i = tid; i < G3_ * H_ / 2; i += 256) wdst[i] = wsrc[i];
  }
  for (int i = tid; i < 16 * H_; i += 256) { sH32[i] = 0.f; sH16[i] = (_Float16)0.f; }
  if (tid < 32) sL[tid] = 0.f;
  __syncthreads();

  int wave = tid >> 5, lane = tid & 31, g = lane >> 4, ln = lane & 15;
  for (int step = 0; step < 63; ++step) {
    int t = dir ? (62 - step) : step;
    // ---- gh = state @ Whh^T via WMMA; each wave owns 3 column tiles (48 cols) ----
    for (int j = 0; j < 3; ++j) {
      int cb = wave * 48 + j * 16;
      v8f acc;
#pragma unroll
      for (int e = 0; e < 8; ++e) acc[e] = 0.f;
#pragma unroll
      for (int kc = 0; kc < 4; ++kc) {
        int kb = kc * 32;
        v8h a0 = *(const v8h*)&sH16[ln * H_ + kb + 16 * g];
        v8h a1 = *(const v8h*)&sH16[ln * H_ + kb + 16 * g + 8];
        v16h a;
#pragma unroll
        for (int e = 0; e < 8; ++e) { a[e] = a0[e]; a[e + 8] = a1[e]; }
        v16h bm = *(const v16h*)&sWhh[(cb + ln) * H_ + kb + 16 * g];
        acc = wmma16x16x32(a, bm, acc);
      }
      // epilogue: gather precomputed input projection gi and combine
#pragma unroll
      for (int v = 0; v < 8; ++v) {
        int m = v + 8 * g;
        int c = cb + ln;
        int row = rb + m;
        int nag = row & 63, bag = row >> 6;
        int om = (t < nag) ? t : t + 1;            // IDX[n][t]
        int grow = (bag << 6) + om;
        float gi = P[(size_t)row * PLD + selfOff + c] + P[(size_t)grow * PLD + otherOff + c];
        float gh = acc[v];
        if (c < 256) sG[m * G3_ + c] = gh + gi + bih[c] + bhh[c];
        else { sG[m * G3_ + c] = gh + bhh[c]; sGin[m * H_ + (c - 256)] = gi + bih[c]; }
      }
    }
    __syncthreads();
    // ---- GRU elementwise update + fused W_hard logit accumulation ----
    for (int idx = tid; idx < 16 * H_; idx += 256) {
      int m = idx >> 7, jj = idx & 127;
      float r = sigmoidf_(sG[m * G3_ + jj]);
      float z = sigmoidf_(sG[m * G3_ + 128 + jj]);
      float nn = tanhf(sGin[idx] + r * sG[m * G3_ + 256 + jj]);
      float hn = (1.f - z) * nn + z * sH32[idx];
      sH32[idx] = hn;
      sH16[idx] = (_Float16)hn;
      atomicAdd(&sL[(m << 1) + 0], hn * W_hard[hOff + jj]);
      atomicAdd(&sL[(m << 1) + 1], hn * W_hard[256 + hOff + jj]);
    }
    __syncthreads();
    if (tid < 32) {
      int m = tid >> 1, jj = tid & 1;
      lout[((size_t)(rb + m) * 63 + t) * 2 + jj] = sL[tid];
      sL[tid] = 0.f;   // next atomics happen after next barrier
    }
  }
}

// ---------------- straight-through gumbel hard weights ----------------
__global__ void hardw_kernel(const float* __restrict__ lf, const float* __restrict__ lb,
                             const float* __restrict__ b_hard, const float* __restrict__ gum,
                             float* __restrict__ hw) {
  int i = blockIdx.x * blockDim.x + threadIdx.x;
  if (i >= BN_ * 63) return;
  float l0 = lf[2 * i]     + lb[2 * i]     + b_hard[0] + gum[2 * i];
  float l1 = lf[2 * i + 1] + lb[2 * i + 1] + b_hard[1] + gum[2 * i + 1];
  hw[i] = (l1 > l0) ? 1.f : 0.f;   // forward value of ST-gumbel == one-hot argmax
}

// ---------------- soft attention + action head (one block per row) ----------------
__global__ void __launch_bounds__(128) attn_kernel(
    const float* __restrict__ P, const float* __restrict__ h,
    const float* __restrict__ hardw, const float* __restrict__ W_act,
    const float* __restrict__ b_act, float* __restrict__ actions, float* __restrict__ alp) {
  int row = blockIdx.x; int b = row >> 6; int n = row & 63; int tid = threadIdx.x;
  __shared__ float sK[64 * 128];
  __shared__ float sHH[64 * 128];
  __shared__ float sQ[128];
  __shared__ float sW[64];
  __shared__ float sR[16][128];
  for (int i = tid; i < 64 * 128; i += 128) {
    int m = i >> 7, c = i & 127;
    sK[i]  = P[(size_t)((b << 6) + m) * PLD + OFF_K + c];
    sHH[i] = h[((b << 6) + m) * H_ + c];
  }
  sQ[tid] = P[(size_t)row * PLD + OFF_Q + tid];
  __syncthreads();
  if (tid < 64) {
    if (tid == n) sW[tid] = 0.f;
    else {
      float d = 0.f;
      for (int c = 0; c < 128; ++c) d += sQ[c] * sK[tid * 128 + c];
      sW[tid] = d * 0.08838834764831845f;  // 1/sqrt(128)
    }
  }
  __syncthreads();
  if (tid == 0) {
    float mx = -3.0e38f;
    for (int m = 0; m < 64; ++m) if (m != n && sW[m] > mx) mx = sW[m];
    float s = 0.f;
    for (int m = 0; m < 64; ++m) if (m != n) { float e = expf(sW[m] - mx); sW[m] = e; s += e; }
    float inv = 1.f / s;
    for (int m = 0; m < 64; ++m) {
      if (m == n) { sW[m] = 0.f; continue; }
      int t = (m < n) ? m : m - 1;
      sW[m] = sW[m] * inv * hardw[(size_t)row * 63 + t];
    }
  }
  __syncthreads();
  float x = 0.f;
  for (int m = 0; m < 64; ++m) x += sHH[(m << 7) + tid] * sW[m];
  float hv = sHH[(n << 7) + tid];
#pragma unroll
  for (int j = 0; j < 16; ++j)
    sR[j][tid] = hv * W_act[j * 256 + tid] + x * W_act[j * 256 + 128 + tid];
  __syncthreads();
  for (int s = 64; s >= 1; s >>= 1) {
    if (tid < s) {
#pragma unroll
      for (int j = 0; j < 16; ++j) sR[j][tid] += sR[j][tid + s];
    }
    __syncthreads();
  }
  if (tid == 0) {
    float lg[16]; float mx = -3.0e38f; int am = 0;
    for (int j = 0; j < 16; ++j) lg[j] = sR[j][0] + b_act[j];
    for (int j = 0; j < 16; ++j) if (lg[j] > mx) { mx = lg[j]; am = j; }
    float s = 0.f;
    for (int j = 0; j < 16; ++j) s += expf(lg[j] - mx);
    alp[row] = lg[am] - (mx + logf(s));
    actions[row] = (float)am;
  }
}

// ---------------- host ----------------
static inline int cdiv(int a, int b) { return (a + b - 1) / b; }

extern "C" void kernel_launch(void* const* d_in, const int* in_sizes, int n_in,
                              void* d_out, int out_size, void* d_ws, size_t ws_size,
                              hipStream_t stream) {
  (void)in_sizes; (void)n_in; (void)out_size; (void)ws_size;
  const float* obs      = (const float*)d_in[0];
  const float* lstm_h   = (const float*)d_in[1];
  const float* lstm_c   = (const float*)d_in[2];
  const float* masks    = (const float*)d_in[3];
  const float* W_enc    = (const float*)d_in[4];
  const float* b_enc    = (const float*)d_in[5];
  const float* lstm_Wih = (const float*)d_in[6];
  const float* lstm_Whh = (const float*)d_in[7];
  const float* lstm_bih = (const float*)d_in[8];
  const float* lstm_bhh = (const float*)d_in[9];
  const float* gWih_f   = (const float*)d_in[10];
  const float* gWhh_f   = (const float*)d_in[11];
  const float* gbih_f   = (const float*)d_in[12];
  const float* gbhh_f   = (const float*)d_in[13];
  const float* gWih_b   = (const float*)d_in[14];
  const float* gWhh_b   = (const float*)d_in[15];
  const float* gbih_b   = (const float*)d_in[16];
  const float* gbhh_b   = (const float*)d_in[17];
  const float* W_hard   = (const float*)d_in[18];
  const float* b_hard   = (const float*)d_in[19];
  const float* Wq       = (const float*)d_in[20];
  const float* Wk       = (const float*)d_in[21];
  const float* W_act    = (const float*)d_in[22];
  const float* b_act    = (const float*)d_in[23];
  const float* gumbel   = (const float*)d_in[24];

  float* out = (float*)d_out;
  float* actions = out;                 // 4096
  float* alp     = out + BN_;           // 4096
  float* hout    = out + 2 * BN_;       // 4096*128 (h_new)
  float* cout    = hout + BN_ * H_;     // 4096*128 (c_out)

  char* ws = (char*)d_ws;
  size_t off = 0;
  auto alloc = [&](size_t bytes) -> char* {
    char* p = ws + off; off = (off + bytes + 255) & ~(size_t)255; return p;
  };
  _Float16* obs16  = (_Float16*)alloc((size_t)BN_ * 128 * 2);
  _Float16* Wenc16 = (_Float16*)alloc((size_t)128 * 128 * 2);
  _Float16* A2     = (_Float16*)alloc((size_t)BN_ * 256 * 2);   // [h_enc | hprev]
  _Float16* W2     = (_Float16*)alloc((size_t)512 * 256 * 2);   // [lstm_Wih | lstm_Whh]
  float*    gates  = (float*)   alloc((size_t)BN_ * 512 * 4);
  _Float16* h16    = (_Float16*)alloc((size_t)BN_ * 128 * 2);
  _Float16* Wcat   = (_Float16*)alloc((size_t)PLD * 128 * 2);   // [Wq|Wk|FS|FO|BS|BO]
  _Float16* whhF16 = (_Float16*)alloc((size_t)G3_ * H_ * 2);
  _Float16* whhB16 = (_Float16*)alloc((size_t)G3_ * H_ * 2);
  float*    P      = (float*)   alloc((size_t)BN_ * PLD * 4);
  float*    logF   = (float*)   alloc((size_t)BN_ * 63 * 2 * 4);
  float*    logB   = (float*)   alloc((size_t)BN_ * 63 * 2 * 4);
  float*    hardw  = (float*)   alloc((size_t)BN_ * 63 * 4);

  auto cvt = [&](const float* src, int sld, _Float16* dst, int dld, int rows, int cols) {
    int ne = rows * cols;
    cvt_f32_f16<<<cdiv(ne, 256), 256, 0, stream>>>(src, sld, dst, dld, rows, cols);
  };
  // weight/activation conversions
  cvt(obs, 128, obs16, 128, BN_, 128);
  cvt(W_enc, 128, Wenc16, 128, 128, 128);
  cvt(lstm_Wih, 128, W2, 256, 512, 128);
  cvt(lstm_Whh, 128, W2 + 128, 256, 512, 128);
  cvt(Wq, 128, Wcat + (size_t)OFF_Q  * 128, 128, 128, 128);
  cvt(Wk, 128, Wcat + (size_t)OFF_K  * 128, 128, 128, 128);
  cvt(gWih_f,       256, Wcat + (size_t)OFF_FS * 128, 128, G3_, 128);
  cvt(gWih_f + 128, 256, Wcat + (size_t)OFF_FO * 128, 128, G3_, 128);
  cvt(gWih_b,       256, Wcat + (size_t)OFF_BS * 128, 128, G3_, 128);
  cvt(gWih_b + 128, 256, Wcat + (size_t)OFF_BO * 128, 128, G3_, 128);
  cvt(gWhh_f, 128, whhF16, 128, G3_, 128);
  cvt(gWhh_b, 128, whhB16, 128, G3_, 128);
  hprev_cvt<<<cdiv(BN_ * H_, 256), 256, 0, stream>>>(lstm_h, masks, A2, 256);

  // encoder: h_enc = relu(obs @ W_enc^T + b_enc) -> A2[:,0:128] (f16)
  {
    int tiles = (BN_ / 16) * (128 / 16);
    gemm_wmma<<<cdiv(tiles, 8), 256, 0, stream>>>(obs16, 128, Wenc16, 128, b_enc, nullptr,
                                                  nullptr, 0, A2, 256, BN_, 128, 128, 1);
  }
  // LSTM gates = [h_enc,hprev] @ [Wih,Whh]^T + bih + bhh
  {
    int tiles = (BN_ / 16) * (512 / 16);
    gemm_wmma<<<cdiv(tiles, 8), 256, 0, stream>>>(A2, 256, W2, 256, lstm_bih, lstm_bhh,
                                                  gates, 512, nullptr, 0, BN_, 512, 256, 0);
  }
  lstm_ew<<<cdiv(BN_ * H_, 256), 256, 0, stream>>>(gates, lstm_c, masks, hout, cout, h16);
  // fused projections: P = h @ [Wq|Wk|FS|FO|BS|BO]^T
  {
    int tiles = (BN_ / 16) * (PLD / 16);
    gemm_wmma<<<cdiv(tiles, 8), 256, 0, stream>>>(h16, 128, Wcat, 128, nullptr, nullptr,
                                                  P, PLD, nullptr, 0, BN_, PLD, 128, 0);
  }
  // bidirectional GRU scan (fused W_hard logits)
  gru_kernel<<<512, 256, 0, stream>>>(P, whhF16, whhB16, gbih_f, gbhh_f, gbih_b, gbhh_b,
                                      W_hard, logF, logB);
  // gumbel straight-through hard weights
  hardw_kernel<<<cdiv(BN_ * 63, 256), 256, 0, stream>>>(logF, logB, b_hard, gumbel, hardw);
  // attention + action head
  attn_kernel<<<BN_, 128, 0, stream>>>(P, hout, hardw, W_act, b_act, actions, alp);
}
